// VanillaAttention_38156489457873
// MI455X (gfx1250) — matrix operand order repaired and checked by reading, not compiled, not run on hardware
//
#include <hip/hip_runtime.h>

typedef _Float16 v16h __attribute__((ext_vector_type(16)));
typedef _Float16 v8h  __attribute__((ext_vector_type(8)));
typedef _Float16 v2h  __attribute__((ext_vector_type(2)));
typedef float    v8f  __attribute__((ext_vector_type(8)));

#define LOG2E 1.44269504088896340736f

// Block-cooperative flash-attention forward (causal), fp32 in/out,
// f16 WMMA matmuls with fp32 accumulation, software-pipelined staging.
// qkv layout: [B][L][3][H][E], out layout: [B][L][H][E]
// 4 waves/block cover 64 consecutive query rows of one (b,h).
// K/V 32-key chunks staged once per block into LDS (f32->f16 at staging);
// next chunk's global loads are issued before compute to hide HBM/L2 latency.
// Row sums of P are computed with an extra WMMA against an all-ones B matrix
// (replaces a 16-lane butterfly reduction per row).
__global__ __launch_bounds__(128, 1) void
fa_fwd_gfx1250(const float* __restrict__ qkv, float* __restrict__ out)
{
    constexpr int L = 2048, H = 16, E = 128;
    constexpr int NW = 4;                        // waves per block

    const int tid    = threadIdx.x;              // 0..127
    const int lane   = tid & 31;
    const int wave   = tid >> 5;

    const int qblk64 = blockIdx.x & (L / 64 - 1);    // 0..31 (64-row block)
    const int bh     = blockIdx.x >> 5;              // 0..31
    const int b      = bh >> 4;
    const int h      = bh & (H - 1);
    const int q0base = qblk64 * 64;
    const int q0     = q0base + wave * 16;           // this wave's query tile
    const int jend_w = q0 + 16;                      // wave's causal key bound
    const int jendmx = q0base + 64;                  // block's max key bound

    const int n_lane = lane & 15;                    // fragment column index
    const int hi     = lane >> 4;                    // lane half-group select

    const size_t rowstride = (size_t)3 * H * E;      // floats between rows l -> l+1
    const float* Qb = qkv + (size_t)b * L * rowstride + (size_t)(0 * H + h) * E;
    const float* Kb = qkv + (size_t)b * L * rowstride + (size_t)(1 * H + h) * E;
    const float* Vb = qkv + (size_t)b * L * rowstride + (size_t)(2 * H + h) * E;

    // ---- LDS: staged f16 K/V chunk + per-wave P transpose buffer ----
    __shared__ _Float16 Ks[32 * 128];        // row-major: half(j_local, e) = j*128 + e
    __shared__ _Float16 Vs[128 * 32];        // transposed: half(e, j_local) = e*32 + j
    __shared__ _Float16 Pbuf[NW][16 * 32];
    _Float16* pb = &Pbuf[wave][0];

    // staging assignment: thread t -> j-pair p = t&15 (rows 2p, 2p+1),
    //                     e-group g = t>>4 (e = 16g .. 16g+15)
    const int sp    = tid & 15;
    const int ebase = (tid >> 4) * 16;

    // pipelined staging registers (raw f32 for K rows 2sp,2sp+1 and V rows)
    float kreg[32], vreg0[16], vreg1[16];

    auto issue_loads = [&](int jc) {
        const float4* kg0 = (const float4*)(Kb + (size_t)(jc + 2 * sp)     * rowstride + ebase);
        const float4* kg1 = (const float4*)(Kb + (size_t)(jc + 2 * sp + 1) * rowstride + ebase);
        const float4* vg0 = (const float4*)(Vb + (size_t)(jc + 2 * sp)     * rowstride + ebase);
        const float4* vg1 = (const float4*)(Vb + (size_t)(jc + 2 * sp + 1) * rowstride + ebase);
        #pragma unroll
        for (int t = 0; t < 4; ++t) {
            ((float4*)kreg)[t]     = kg0[t];
            ((float4*)kreg)[4 + t] = kg1[t];
            ((float4*)vreg0)[t]    = vg0[t];
            ((float4*)vreg1)[t]    = vg1[t];
        }
    };

    auto store_lds = [&]() {
        #pragma unroll
        for (int rr = 0; rr < 2; ++rr) {
            v8h h0, h1;
            #pragma unroll
            for (int i = 0; i < 8; ++i) {
                h0[i] = (_Float16)kreg[16 * rr + i];
                h1[i] = (_Float16)kreg[16 * rr + 8 + i];
            }
            *(v8h*)&Ks[(2 * sp + rr) * 128 + ebase]     = h0;   // ds_store_b128
            *(v8h*)&Ks[(2 * sp + rr) * 128 + ebase + 8] = h1;   // ds_store_b128
        }
        #pragma unroll
        for (int i = 0; i < 16; ++i) {
            v2h w; w[0] = (_Float16)vreg0[i]; w[1] = (_Float16)vreg1[i];
            *(v2h*)&Vs[(ebase + i) * 32 + 2 * sp] = w;          // ds_store_b32
        }
    };

    // ---------------- load Q tile as 4 A-fragments (16x32 f16) ----------------
    // A-layout: lane holds M = lane&15; half i -> K = 32c + hi*16 + i
    const float qscale = 0.08838834764831845f * LOG2E;   // 1/sqrt(E) * log2(e)
    v16h qa[4];
    {
        const float* qrow = Qb + (size_t)(q0 + n_lane) * rowstride;
        #pragma unroll
        for (int c = 0; c < 4; ++c) {
            #pragma unroll
            for (int i = 0; i < 16; ++i) {
                const int e = 32 * c + hi * 16 + i;
                qa[c][i] = (_Float16)(qrow[e] * qscale);
            }
        }
    }

    // all-ones B fragment for WMMA row-sum reduction
    v16h ones;
    #pragma unroll
    for (int i = 0; i < 16; ++i) ones[i] = (_Float16)1.0f;

    // ---------------- accumulators + online-softmax state ----------------
    v8f o[8];
    #pragma unroll
    for (int f = 0; f < 8; ++f)
        #pragma unroll
        for (int r = 0; r < 8; ++r) o[f][r] = 0.0f;

    float mrow[8], lrow[8];
    #pragma unroll
    for (int r = 0; r < 8; ++r) { mrow[r] = -1.0e30f; lrow[r] = 0.0f; }

    issue_loads(0);                               // pipeline prologue

    for (int j0 = 0; j0 < jendmx; j0 += 32) {
        // ---- commit staged chunk to LDS (f32->f16 conversion happens here) ----
        store_lds();
        __syncthreads();

        // ---- issue next chunk's global loads: latency hides behind compute ----
        if (j0 + 32 < jendmx) issue_loads(j0 + 32);
        if (j0 + 64 < L)
            __builtin_prefetch(Kb + (size_t)(j0 + 64) * rowstride, 0, 1);

        // ================= per-wave compute (skip past this wave's bound) ======
        if (j0 < jend_w) {
            // ---------- S = Q * K^T : two 16-key C fragments ----------
            v8f s[2];
            #pragma unroll
            for (int nb = 0; nb < 2; ++nb) {
                v8f acc;
                #pragma unroll
                for (int r = 0; r < 8; ++r) acc[r] = 0.0f;

                const _Float16* krow = &Ks[(nb * 16 + n_lane) * 128];
                #pragma unroll
                for (int c = 0; c < 4; ++c) {
                    // B-layout: half i -> K-dim = 32c + hi*16 + i (contiguous in LDS)
                    const v8h* kp = (const v8h*)(krow + 32 * c + hi * 16);
                    const v8h  k0 = kp[0];                   // ds_load_b128
                    const v8h  k1 = kp[1];                   // ds_load_b128
                    const v16h kb = __builtin_shufflevector(k0, k1,
                        0, 1, 2, 3, 4, 5, 6, 7, 8, 9, 10, 11, 12, 13, 14, 15);
                    acc = __builtin_amdgcn_wmma_f32_16x16x32_f16(
                              false, qa[c], false, kb, (short)0, acc, false, false);
                }
                s[nb] = acc;
            }

            // ---------- causal mask (only the diagonal block) ----------
            if (j0 + 32 >= jend_w) {
                #pragma unroll
                for (int nb = 0; nb < 2; ++nb) {
                    const int key = j0 + nb * 16 + n_lane;
                    #pragma unroll
                    for (int r = 0; r < 8; ++r) {
                        const int row = q0 + hi * 8 + r;
                        if (key > row) s[nb][r] = -1.0e30f;
                    }
                }
            }

            // ---------- online softmax: max butterfly + exp (base-2) ----------
            float alpha[8];
            #pragma unroll
            for (int r = 0; r < 8; ++r) {
                float lm = fmaxf(s[0][r], s[1][r]);
                #pragma unroll
                for (int off = 1; off < 16; off <<= 1)
                    lm = fmaxf(lm, __shfl_xor(lm, off, 32));   // stays in 16-lane group
                const float mnew = fmaxf(mrow[r], lm);
                alpha[r] = exp2f(mrow[r] - mnew);
                s[0][r] = exp2f(s[0][r] - mnew);
                s[1][r] = exp2f(s[1][r] - mnew);
                mrow[r] = mnew;
            }
            #pragma unroll
            for (int f = 0; f < 8; ++f)
                #pragma unroll
                for (int r = 0; r < 8; ++r) o[f][r] *= alpha[r];

            // ---------- P: C-layout -> per-wave LDS -> A-layout ----------
            #pragma unroll
            for (int r = 0; r < 8; ++r) {
                const int m = hi * 8 + r;
                pb[m * 32 + n_lane]      = (_Float16)s[0][r];
                pb[m * 32 + 16 + n_lane] = (_Float16)s[1][r];
            }
            // LDS is in-order within a wave; also a compiler barrier.
            asm volatile("s_wait_dscnt 0" ::: "memory");

            v16h pa;
            #pragma unroll
            for (int i = 0; i < 16; ++i) {
                const int j = hi * 16 + i;  // A-layout K index
                pa[i] = pb[n_lane * 32 + j];
            }

            // ---------- row sums via WMMA against all-ones B ----------
            v8f sumf;
            #pragma unroll
            for (int r = 0; r < 8; ++r) sumf[r] = 0.0f;
            sumf = __builtin_amdgcn_wmma_f32_16x16x32_f16(
                       false, pa, false, ones, (short)0, sumf, false, false);
            #pragma unroll
            for (int r = 0; r < 8; ++r)
                lrow[r] = lrow[r] * alpha[r] + sumf[r];

            // ---------- O += P * V : 8 WMMAs covering E = 128 ----------
            #pragma unroll
            for (int ne = 0; ne < 8; ++ne) {
                // B-layout: half i -> key j = hi*16 + i ; column e = ne*16 + n_lane
                // Vs transposed layout makes this contiguous.
                const v8h* vp = (const v8h*)&Vs[(ne * 16 + n_lane) * 32 + hi * 16];
                const v8h  w0 = vp[0];                        // ds_load_b128
                const v8h  w1 = vp[1];                        // ds_load_b128
                const v16h vb = __builtin_shufflevector(w0, w1,
                    0, 1, 2, 3, 4, 5, 6, 7, 8, 9, 10, 11, 12, 13, 14, 15);
                o[ne] = __builtin_amdgcn_wmma_f32_16x16x32_f16(
                            false, pa, false, vb, (short)0, o[ne], false, false);
            }
        }
        __syncthreads();   // all waves done reading Ks/Vs before next staging
    }

    // ---------------- finalize: divide by row sums, write out ----------------
    #pragma unroll
    for (int r = 0; r < 8; ++r) {
        const float rinv = 1.0f / lrow[r];
        const int   row  = q0 + hi * 8 + r;
        float* op = out + (((size_t)b * L + row) * H + h) * E;
        #pragma unroll
        for (int ne = 0; ne < 8; ++ne)
            op[ne * 16 + n_lane] = o[ne][r] * rinv;
    }
}

extern "C" void kernel_launch(void* const* d_in, const int* in_sizes, int n_in,
                              void* d_out, int out_size, void* d_ws, size_t ws_size,
                              hipStream_t stream) {
    (void)in_sizes; (void)n_in; (void)out_size; (void)d_ws; (void)ws_size;
    const float* qkv = (const float*)d_in[0];
    float*       out = (float*)d_out;

    constexpr int B = 2, L = 2048, H = 16;
    const int blocks = B * H * (L / 64);   // 1024 blocks, 4 waves each
    fa_fwd_gfx1250<<<blocks, 128, 0, stream>>>(qkv, out);
}
